// SelfAttentionHead_59777354826246
// MI455X (gfx1250) — matrix operand order repaired and checked by reading, not compiled, not run on hardware
//
#include <hip/hip_runtime.h>

#define BATCH 8
#define SEQ   2048
#define EMBED 1024
#define HEAD  64

typedef __attribute__((ext_vector_type(16))) _Float16 v16h;
typedef __attribute__((ext_vector_type(8)))  float    v8f;

union Frag16 {
  v16h h;
  unsigned int u[8];
};

__device__ __forceinline__ v8f wmma_f16(v16h a, v16h b, v8f c) {
  // D = A(16x32 f16) x B(32x16 f16) + C(16x16 f32)
  return __builtin_amdgcn_wmma_f32_16x16x32_f16(false, a, false, b, (short)0, c,
                                                false, false);
}

// A-fragment (16x32 f16) from row-major f16 memory.
// Layout: lane L<16 -> row M=L, K=0..15 (v0..7);
//         lane L>=16 -> row M=L-16, K=16..31. Pairs packed per VGPR.
// Each half is 8 consecutive f16 (16B) -> compiler fuses to global_load_b128.
__device__ __forceinline__ v16h load_afrag_f16(const _Float16* base, int stride,
                                               int lane) {
  const int m   = lane & 15;
  const int kb8 = (lane >> 4) << 3;
  Frag16 f;
#pragma unroll
  for (int v = 0; v < 8; ++v) {
    const int K = 2 * kb8 + 2 * v;
    f.u[v] = *reinterpret_cast<const unsigned int*>(base + (size_t)m * stride + K);
  }
  return f.h;
}

// B-fragment (32x16 f16) loaded from B-transposed row-major memory (rows = N,
// cols = K) so each packed (K,K+1) pair is contiguous.
// Layout: lane L<16 -> col N=L, K=0..15 ; lane L>=16 -> col N=L-16, K=16..31.
__device__ __forceinline__ v16h load_bfrag_f16(const _Float16* baseT, int stride,
                                               int lane) {
  const int n    = lane & 15;
  const int kb16 = (lane >> 4) << 4;
  Frag16 f;
#pragma unroll
  for (int v = 0; v < 8; ++v) {
    f.u[v] = *reinterpret_cast<const unsigned int*>(baseT + (size_t)n * stride +
                                                    kb16 + 2 * v);
  }
  return f.h;
}

// Exchange a value with lane^16 (wave32): ds_swizzle BITMASK mode,
// offset = xor[14:10]|or[9:5]|and[4:0] -> SWAPX16 = (0x10<<10)|0x1F.
__device__ __forceinline__ float swap16(float v) {
  return __builtin_bit_cast(
      float, __builtin_amdgcn_ds_swizzle(__builtin_bit_cast(int, v), 0x401F));
}

// ---------------------------------------------------------------------------
// Phase 0: Wq/Wk/Wv [C][H] f32  ->  WT [3][H][C] f16 (transposed, f16)
// ---------------------------------------------------------------------------
__global__ void __launch_bounds__(256) wt_transpose_kernel(
    const float* __restrict__ Wq, const float* __restrict__ Wk,
    const float* __restrict__ Wv, _Float16* __restrict__ WT) {
  const int id  = blockIdx.x * 256 + threadIdx.x;  // 0 .. 3*H*C-1
  const int w   = id / (HEAD * EMBED);
  const int rem = id - w * (HEAD * EMBED);
  const int n   = rem >> 10;    // / EMBED
  const int c   = rem & (EMBED - 1);
  const float* W = (w == 0) ? Wq : ((w == 1) ? Wk : Wv);
  WT[id] = (_Float16)W[(size_t)c * HEAD + n];
}

// ---------------------------------------------------------------------------
// Phase 1: q/k/v projection. One block = 16 rows of x; 4 waves, each wave owns
// a 16-wide slice of HEAD and accumulates q,k,v via WMMA over K=1024.
// Outputs: q,k row-major [B][T][H] f16 ; v transposed [B][H][T] f16.
// ---------------------------------------------------------------------------
__global__ void __launch_bounds__(128) qkv_proj_kernel(
    const float* __restrict__ x, const _Float16* __restrict__ WT,
    _Float16* __restrict__ qo, _Float16* __restrict__ ko,
    _Float16* __restrict__ vTo) {
  const int tile = blockIdx.x;          // B*SEQ/16 tiles
  const int b    = tile >> 7;           // SEQ/16 = 128 tiles per batch
  const int t0   = (tile & 127) << 4;
  const int lane = threadIdx.x & 31;
  const int wave = threadIdx.x >> 5;
  const int n0   = wave << 4;

  const float* xt = x + ((size_t)b * SEQ + t0) * EMBED;
  const _Float16* WTq = WT;
  const _Float16* WTk = WT + (size_t)HEAD * EMBED;
  const _Float16* WTv = WT + (size_t)2 * HEAD * EMBED;

  const int m   = lane & 15;
  const int kb8 = (lane >> 4) << 3;

  v8f aq = {}, ak = {}, av = {};
  for (int kk = 0; kk < EMBED; kk += 32) {
    // Build A fragment from f32 x, converting to f16 in-register.
    Frag16 a;
#pragma unroll
    for (int v = 0; v < 8; ++v) {
      const int K = 2 * kb8 + 2 * v;
      const float2 p =
          *reinterpret_cast<const float2*>(xt + (size_t)m * EMBED + kk + K);
      const unsigned short lo = __builtin_bit_cast(unsigned short, (_Float16)p.x);
      const unsigned short hi = __builtin_bit_cast(unsigned short, (_Float16)p.y);
      a.u[v] = (unsigned int)lo | ((unsigned int)hi << 16);
    }
    const v16h bq = load_bfrag_f16(WTq + (size_t)n0 * EMBED + kk, EMBED, lane);
    const v16h bk = load_bfrag_f16(WTk + (size_t)n0 * EMBED + kk, EMBED, lane);
    const v16h bv = load_bfrag_f16(WTv + (size_t)n0 * EMBED + kk, EMBED, lane);
    aq = wmma_f16(a.h, bq, aq);
    ak = wmma_f16(a.h, bk, ak);
    av = wmma_f16(a.h, bv, av);
  }

  // C/D layout store: VGPR r -> row r + 8*(lane>=16), col = lane&15.
  const int n = lane & 15;
#pragma unroll
  for (int r = 0; r < 8; ++r) {
    const int mr = r + kb8;
    const size_t idx = ((size_t)b * SEQ + t0 + mr) * HEAD + n0 + n;
    qo[idx] = (_Float16)aq[r];
    ko[idx] = (_Float16)ak[r];
    vTo[((size_t)b * HEAD + n0 + n) * SEQ + t0 + mr] = (_Float16)av[r];
  }
}

// ---------------------------------------------------------------------------
// Phase 2: causal flash attention, transposed-score formulation.
//   S' = K · Qᵀ  (so each lane owns ONE query; row stats are in-lane)
//   Oᵀ = Vᵀ · Pᵀ (Pᵀ already in registers; 8 swizzles -> B fragment)
// One wave per 16-query tile, 32 keys per iteration. No LDS memory traffic.
// ---------------------------------------------------------------------------
__global__ void __launch_bounds__(128) attn_kernel(
    const _Float16* __restrict__ q, const _Float16* __restrict__ k,
    const _Float16* __restrict__ vT, float* __restrict__ out) {
  const int lane = threadIdx.x & 31;
  const int wave = threadIdx.x >> 5;
  const int wq   = blockIdx.x * 4 + wave;  // 0 .. B*SEQ/16 - 1
  const int b    = wq >> 7;
  const int q0   = (wq & 127) << 4;

  const _Float16* qb  = q  + (size_t)b * SEQ * HEAD;
  const _Float16* kbp = k  + (size_t)b * SEQ * HEAD;
  const _Float16* vb  = vT + (size_t)b * HEAD * SEQ;

  // Q as resident B-fragments (Qᵀ operand), head 0..31 and 32..63.
  const v16h qbf0 = load_bfrag_f16(qb + (size_t)q0 * HEAD + 0,  HEAD, lane);
  const v16h qbf1 = load_bfrag_f16(qb + (size_t)q0 * HEAD + 32, HEAD, lane);

  // Oᵀ accumulators: tile t = heads t*16..t*16+15; lane n = query.
  v8f o0 = {}, o1 = {}, o2 = {}, o3 = {};
  float mrun = -1e30f, lrun = 0.0f;   // per-lane (per-query) softmax state

  const int n    = lane & 15;
  const int h8   = (lane >> 4) << 3;  // 0 or 8: key sub-row offset in D layout
  const bool hiH = (lane >= 16);
  // scores * C^-0.5, folded with log2(e) so we can softmax in base-2.
  const float ksc = 0.03125f * 1.4426950408889634f;
  const int query = q0 + n;

  const int nkb = (q0 + 16 + 31) >> 5;  // causal: key blocks touching this tile
  for (int kbi = 0; kbi < nkb; ++kbi) {
    const int kt0 = kbi << 5;

    if (kbi + 1 < nkb) {  // wave-uniform: prefetch next K/V block
      const int nt = kt0 + 32;
      __builtin_prefetch(kbp + (size_t)nt * HEAD + (size_t)lane * 64, 0, 3);
      __builtin_prefetch(vb + (size_t)lane * SEQ + nt, 0, 3);
      __builtin_prefetch(vb + (size_t)(lane + 32) * SEQ + nt, 0, 3);
    }

    // S'[key][query]: two 16x16 tiles (keys kt0..+15 and kt0+16..+31).
    v8f s0 = {}, s1 = {};
    s0 = wmma_f16(load_afrag_f16(kbp + (size_t)(kt0)      * HEAD + 0,  HEAD, lane), qbf0, s0);
    s0 = wmma_f16(load_afrag_f16(kbp + (size_t)(kt0)      * HEAD + 32, HEAD, lane), qbf1, s0);
    s1 = wmma_f16(load_afrag_f16(kbp + (size_t)(kt0 + 16) * HEAD + 0,  HEAD, lane), qbf0, s1);
    s1 = wmma_f16(load_afrag_f16(kbp + (size_t)(kt0 + 16) * HEAD + 32, HEAD, lane), qbf1, s1);

    // Scale + causal mask. Element r of s0: key = kt0+r+h8 ; of s1: +16.
    float sv[16];
#pragma unroll
    for (int r = 0; r < 8; ++r) {
      const int key0 = kt0 + r + h8;
      sv[r]     = (key0      > query) ? -1e30f : s0[r] * ksc;
      sv[8 + r] = (key0 + 16 > query) ? -1e30f : s1[r] * ksc;
    }

    // Row (= per-query) max: in-lane over 16 regs + one lane^16 exchange.
    float lmax = sv[0];
#pragma unroll
    for (int j = 1; j < 16; ++j) lmax = fmaxf(lmax, sv[j]);
    const float rmax  = fmaxf(lmax, swap16(lmax));
    const float mnew  = fmaxf(mrun, rmax);
    const float alpha = exp2f(mrun - mnew);
    mrun = mnew;

    float p[16];
    float lsum = 0.0f;
#pragma unroll
    for (int j = 0; j < 16; ++j) {
      p[j] = exp2f(sv[j] - mnew);
      lsum += p[j];
    }
    lrun = lrun * alpha + lsum + swap16(lsum);

    // Rescale output accumulators (alpha is a per-lane scalar).
#pragma unroll
    for (int r = 0; r < 8; ++r) {
      o0[r] *= alpha; o1[r] *= alpha; o2[r] *= alpha; o3[r] *= alpha;
    }

    // Build Pᵀ B-fragment. Lane holds keys {kt0+h8+r} (p[0..7]) and
    // {kt0+16+h8+r} (p[8..15]). B-frag wants keys 0..15 in lanes<16 and
    // 16..31 in lanes>=16 (pairs (2v,2v+1) per VGPR). One SWAPX16 per r
    // moves the tile the other half needs; selects place lo/hi groups.
    float lo8[8], hi8[8];
#pragma unroll
    for (int r = 0; r < 8; ++r) {
      const float tsend = hiH ? p[r] : p[8 + r];
      const float trecv = swap16(tsend);
      lo8[r] = hiH ? trecv : p[r];       // rel keys 0..7  (or 16..23)
      hi8[r] = hiH ? p[8 + r] : trecv;   // rel keys 8..15 (or 24..31)
    }
    Frag16 pb;
#pragma unroll
    for (int v = 0; v < 4; ++v) {
      const unsigned short a0 = __builtin_bit_cast(unsigned short, (_Float16)lo8[2 * v]);
      const unsigned short a1 = __builtin_bit_cast(unsigned short, (_Float16)lo8[2 * v + 1]);
      const unsigned short b0 = __builtin_bit_cast(unsigned short, (_Float16)hi8[2 * v]);
      const unsigned short b1 = __builtin_bit_cast(unsigned short, (_Float16)hi8[2 * v + 1]);
      pb.u[v]     = (unsigned int)a0 | ((unsigned int)a1 << 16);
      pb.u[4 + v] = (unsigned int)b0 | ((unsigned int)b1 << 16);
    }

    // Oᵀ += Vᵀ(16 heads x 32 keys) · Pᵀ(32 keys x 16 queries), 4 head tiles.
    o0 = wmma_f16(load_afrag_f16(vb + (size_t)0  * SEQ + kt0, SEQ, lane), pb.h, o0);
    o1 = wmma_f16(load_afrag_f16(vb + (size_t)16 * SEQ + kt0, SEQ, lane), pb.h, o1);
    o2 = wmma_f16(load_afrag_f16(vb + (size_t)32 * SEQ + kt0, SEQ, lane), pb.h, o2);
    o3 = wmma_f16(load_afrag_f16(vb + (size_t)48 * SEQ + kt0, SEQ, lane), pb.h, o3);
  }

  // Epilogue: divide by row sum; lane stores 8 contiguous f32 per head tile.
  const float inv = 1.0f / lrun;
  float* orow = out + ((size_t)b * SEQ + query) * HEAD + h8;
#pragma unroll
  for (int r = 0; r < 8; ++r) {
    orow[r]      = o0[r] * inv;
    orow[16 + r] = o1[r] * inv;
    orow[32 + r] = o2[r] * inv;
    orow[48 + r] = o3[r] * inv;
  }
}

// ---------------------------------------------------------------------------
extern "C" void kernel_launch(void* const* d_in, const int* in_sizes, int n_in,
                              void* d_out, int out_size, void* d_ws,
                              size_t ws_size, hipStream_t stream) {
  (void)in_sizes; (void)n_in; (void)out_size; (void)ws_size;

  const float* x  = (const float*)d_in[0];
  const float* Wq = (const float*)d_in[1];
  const float* Wk = (const float*)d_in[2];
  const float* Wv = (const float*)d_in[3];
  float* out = (float*)d_out;

  // Workspace: q (2MB) | k (2MB) | vT (2MB) | WT (384KB), all f16.
  char* ws = (char*)d_ws;
  const size_t QSZ = (size_t)BATCH * SEQ * HEAD * 2;
  _Float16* qb = (_Float16*)(ws);
  _Float16* kb = (_Float16*)(ws + QSZ);
  _Float16* vT = (_Float16*)(ws + 2 * QSZ);
  _Float16* WT = (_Float16*)(ws + 3 * QSZ);

  wt_transpose_kernel<<<(3 * HEAD * EMBED) / 256, 256, 0, stream>>>(Wq, Wk, Wv, WT);
  qkv_proj_kernel<<<BATCH * SEQ / 16, 128, 0, stream>>>(x, WT, qb, kb, vT);
  attn_kernel<<<BATCH * SEQ / 64, 128, 0, stream>>>(qb, kb, vT, out);
}
